// ProteinLigandAttention_28656021799367
// MI455X (gfx1250) — matrix operand order repaired and checked by reading, not compiled, not run on hardware
//
#include <hip/hip_runtime.h>
#include <hip/hip_bf16.h>

#define DD 256
#define NB 256
#define N_ITERS 3
#define AMAX 64
#define RMAX 512
#define SLACK 512

typedef __attribute__((ext_vector_type(16))) __bf16 v16bf;
typedef __attribute__((ext_vector_type(8)))  __bf16 v8bf;
typedef __attribute__((ext_vector_type(8)))  float  v8f;

__device__ __forceinline__ float leaky1(float x) { return x >= 0.f ? x : 0.1f * x; }

__device__ __forceinline__ v8f wmma_acc(v16bf a, v16bf b, v8f c) {
    return __builtin_amdgcn_wmma_f32_16x16x32_bf16(
        /*neg_a=*/false, a, /*neg_b=*/false, b,
        /*c_mod=*/(short)0, c, /*reuse_a=*/false, /*reuse_b=*/false);
}

// A-fragment (16x32 bf16) from row-major bf16, row stride ld elements.
// lane: m=lane&15, half=lane>>4. e<8 -> K = e + 8*half ; e>=8 -> K = (e-8) + 16 + 8*half
__device__ __forceinline__ v16bf load_frag_a(const __bf16* base, int ld, int lane) {
    const int m = lane & 15, h = lane >> 4;
    const __bf16* p = base + m * ld + h * 8;
    const v8bf lo = *(const v8bf*)p;
    const v8bf hi = *(const v8bf*)(p + 16);
    v16bf r;
#pragma unroll
    for (int i = 0; i < 8; ++i) { r[i] = lo[i]; r[i + 8] = hi[i]; }
    return r;
}

// A-fragment from f32 source with row clamp (for reading raw inputs safely).
__device__ __forceinline__ v16bf load_frag_a_f32(const float* X, int ld, int row0,
                                                 int maxrow, int k0, int lane) {
    const int m = lane & 15, h = lane >> 4;
    int r = row0 + m; if (r >= maxrow) r = maxrow - 1; if (r < 0) r = 0;
    const float* p = X + (size_t)r * ld + k0 + h * 8;
    v16bf v;
#pragma unroll
    for (int i = 0; i < 8; ++i) { v[i] = (__bf16)p[i]; v[i + 8] = (__bf16)p[16 + i]; }
    return v;
}

// B-fragment (32x16 bf16) from "transposed" storage bt[n][k], row stride ld.
// lane: n=lane&15, half=lane>>4. e<8 -> K = e + 8*half ; e>=8 -> K = (e-8) + 16 + 8*half
__device__ __forceinline__ v16bf load_frag_b(const __bf16* bt, int ld, int lane) {
    const int n = lane & 15, h = lane >> 4;
    const __bf16* p = bt + (size_t)n * ld + h * 8;
    const v8bf lo = *(const v8bf*)p;
    const v8bf hi = *(const v8bf*)(p + 16);
    v16bf r;
#pragma unroll
    for (int i = 0; i < 8; ++i) { r[i] = lo[i]; r[i + 8] = hi[i]; }
    return r;
}

struct WP { const float* w[10]; };

// Convert ten 256x256 f32 weights into bf16 *transposed* (Wt[n][k] = W[k][n]).
__global__ void wconv_kernel(WP wp, __bf16* __restrict__ wt) {
    const int w = blockIdx.x >> 8;
    const int n = blockIdx.x & 255;
    const int k = threadIdx.x;
    wt[(size_t)w * 65536 + (size_t)n * DD + k] = (__bf16)wp.w[w][(size_t)k * DD + n];
}

// Y = leaky(X @ W) : f32 -> d_out region + bf16 mirror. X is f32 row-major.
__global__ __launch_bounds__(128)
void transform_kernel(const float* __restrict__ X, const __bf16* __restrict__ Wt,
                      float* __restrict__ Yf, __bf16* __restrict__ Ybf, int ntok) {
    const int lane = threadIdx.x & 31, wv = threadIdx.x >> 5;
    const int row0 = blockIdx.x * 16;
    const int n = lane & 15, h = lane >> 4;

    v16bf afr[8];                       // hoist: one load+convert of A per wave
#pragma unroll
    for (int kf = 0; kf < 8; ++kf)
        afr[kf] = load_frag_a_f32(X, DD, row0, ntok, kf * 32, lane);

#pragma unroll
    for (int j = 0; j < 4; ++j) {
        const int n0 = wv * 64 + j * 16;
        v8f acc = {};
#pragma unroll
        for (int kf = 0; kf < 8; ++kf) {
            v16bf b = load_frag_b(Wt + (size_t)n0 * DD + kf * 32, DD, lane);
            acc = wmma_acc(afr[kf], b, acc);
        }
#pragma unroll
        for (int i = 0; i < 8; ++i) {
            const int r = row0 + h * 8 + i;
            if (r < ntok) {
                const float y = leaky1(acc[i]);
                Yf[(size_t)r * DD + n0 + n] = y;
                Ybf[(size_t)r * DD + n0 + n] = (__bf16)y;
            }
        }
    }
}

// Per-complex projections: q,k row-major bf16; v transposed vt[d][b*cmax+tok].
__global__ __launch_bounds__(128)
void proj_kernel(const __bf16* __restrict__ Xbf,
                 const __bf16* __restrict__ Wq, const __bf16* __restrict__ Wk,
                 const __bf16* __restrict__ Wv,
                 __bf16* __restrict__ q, __bf16* __restrict__ kk, __bf16* __restrict__ vt,
                 const long long* __restrict__ scope, int cmax, long long vt_stride) {
    const int b = blockIdx.x;
    const int which = blockIdx.y;
    const long long start = scope[2 * b];
    const int cnt = (int)scope[2 * b + 1];
    const __bf16* W = (which == 0) ? Wq : (which == 1) ? Wk : Wv;
    const int lane = threadIdx.x & 31, wv = threadIdx.x >> 5;
    const int n = lane & 15, h = lane >> 4;

    for (int t0 = 0; t0 < cnt; t0 += 16) {
        const __bf16* xrow = Xbf + (size_t)(start + t0) * DD;
        v16bf afr[8];                   // hoist A fragments across the 4 N-tiles
#pragma unroll
        for (int kf = 0; kf < 8; ++kf)
            afr[kf] = load_frag_a(xrow + kf * 32, DD, lane);

#pragma unroll
        for (int j = 0; j < 4; ++j) {
            const int n0 = wv * 64 + j * 16;
            v8f acc = {};
#pragma unroll
            for (int kf = 0; kf < 8; ++kf) {
                v16bf bb = load_frag_b(W + (size_t)n0 * DD + kf * 32, DD, lane);
                acc = wmma_acc(afr[kf], bb, acc);
            }
            if (which < 2) {
                __bf16* out = (which == 0) ? q : kk;
#pragma unroll
                for (int i = 0; i < 8; ++i) {
                    const int loc = t0 + h * 8 + i;
                    if (loc < cnt)
                        out[(size_t)(start + loc) * DD + n0 + n] = (__bf16)acc[i];
                }
            } else {
#pragma unroll
                for (int i = 0; i < 8; ++i) {
                    const int loc = t0 + h * 8 + i;
                    if (loc < cnt)
                        vt[(size_t)(n0 + n) * vt_stride + (size_t)b * cmax + loc] = (__bf16)acc[i];
                }
            }
        }
    }
}

// Cross attention + out-projection + residual, one complex per block (4 waves).
// X_new[q,:] = leaky( (sigmoid(Q K^T) * mask) @ V @ Wo ) / kc + X_old[q,:]
__global__ __launch_bounds__(128, 1)
void att_kernel(const __bf16* __restrict__ q, const __bf16* __restrict__ k,
                const __bf16* __restrict__ vt, const __bf16* __restrict__ Wo,
                float* __restrict__ X, __bf16* __restrict__ Xbf,
                const long long* __restrict__ qscope, const long long* __restrict__ kscope,
                int cmax, long long vt_stride) {
    __shared__ __align__(16) __bf16 s_lds[4][16 * 32];
    __shared__ __align__(16) __bf16 c_lds[4][16 * 256];
    const int b = blockIdx.x;
    const long long qs = qscope[2 * b];
    const int qc = (int)qscope[2 * b + 1];
    const long long ks = kscope[2 * b];
    const int kc = (int)kscope[2 * b + 1];
    const int lane = threadIdx.x & 31, wv = threadIdx.x >> 5;
    const int n = lane & 15, h = lane >> 4;
    const float inv = 1.0f / (float)kc;
    const __bf16* vtb = vt + (size_t)b * cmax;
    const v8f vzero = {};

    for (int qt = wv * 16; qt < qc; qt += 64) {
        const __bf16* qrow = q + (size_t)(qs + qt) * DD;
        v8f ctx[16];
#pragma unroll
        for (int i = 0; i < 16; ++i) ctx[i] = vzero;

        for (int kch = 0; kch < kc; kch += 32) {
            if (kch + 32 < kc)          // prefetch next chunk's K rows (one line/lane)
                __builtin_prefetch((const void*)(k + (size_t)(ks + kch + 32 + lane) * DD), 0, 0);

            // --- score tile 16x32: one Q fragment feeds both 16-col halves ---
            v8f s0 = vzero, s1 = vzero;
#pragma unroll
            for (int kf = 0; kf < 8; ++kf) {
                v16bf a  = load_frag_a(qrow + kf * 32, DD, lane);
                v16bf b0 = load_frag_b(k + (size_t)(ks + kch) * DD + kf * 32, DD, lane);
                v16bf b1 = load_frag_b(k + (size_t)(ks + kch + 16) * DD + kf * 32, DD, lane);
                s0 = wmma_acc(a, b0, s0);
                s1 = wmma_acc(a, b1, s1);
            }
            const bool val0 = (kch + n) < kc;
            const bool val1 = (kch + 16 + n) < kc;
#pragma unroll
            for (int i = 0; i < 8; ++i) {
                const float f0 = val0 ? (1.0f / (1.0f + __expf(-s0[i]))) : 0.0f;
                const float f1 = val1 ? (1.0f / (1.0f + __expf(-s1[i]))) : 0.0f;
                s_lds[wv][(h * 8 + i) * 32 + n]      = (__bf16)f0;
                s_lds[wv][(h * 8 + i) * 32 + 16 + n] = (__bf16)f1;
            }
            asm volatile("" ::: "memory"); // keep LDS store->load order (DS in-order per wave)

            // --- ctx += S @ V (V pre-transposed -> contiguous B-fragments) ---
            const v16bf a_s = load_frag_a(&s_lds[wv][0], 32, lane);
#pragma unroll
            for (int nt = 0; nt < 16; ++nt) {
                v16bf bv = load_frag_b(vtb + (size_t)(nt * 16) * vt_stride + kch,
                                       (int)vt_stride, lane);
                ctx[nt] = wmma_acc(a_s, bv, ctx[nt]);
            }
        }

        // --- dump ctx (16x256) to LDS as bf16 for re-fragmentation ---
#pragma unroll
        for (int nt = 0; nt < 16; ++nt)
#pragma unroll
            for (int i = 0; i < 8; ++i)
                c_lds[wv][(h * 8 + i) * 256 + nt * 16 + n] = (__bf16)ctx[nt][i];
        asm volatile("" ::: "memory");

        // --- out-projection: hoist the 8 ctx fragments (ctx regs are dead now) ---
        v16bf cfr[8];
#pragma unroll
        for (int kf = 0; kf < 8; ++kf)
            cfr[kf] = load_frag_a(&c_lds[wv][kf * 32], 256, lane);

#pragma unroll
        for (int nt = 0; nt < 16; ++nt) {
            v8f acc = vzero;
#pragma unroll
            for (int kf = 0; kf < 8; ++kf) {
                v16bf bb = load_frag_b(Wo + (size_t)(nt * 16) * DD + kf * 32, DD, lane);
                acc = wmma_acc(cfr[kf], bb, acc);
            }
#pragma unroll
            for (int i = 0; i < 8; ++i) {
                const int loc = qt + h * 8 + i;
                if (loc < qc) {
                    const size_t off = (size_t)(qs + loc) * DD + nt * 16 + n;
                    const float y = leaky1(acc[i]) * inv + X[off];
                    X[off] = y;
                    Xbf[off] = (__bf16)y;
                }
            }
        }
    }
}

extern "C" void kernel_launch(void* const* d_in, const int* in_sizes, int n_in,
                              void* d_out, int out_size, void* d_ws, size_t ws_size,
                              hipStream_t stream) {
    const float* fatoms = (const float*)d_in[0];
    const float* fres   = (const float*)d_in[1];
    const long long* lig = (const long long*)d_in[12];
    const long long* res = (const long long*)d_in[13];
    const int nA = in_sizes[0] / DD;
    const int nR = in_sizes[1] / DD;

    float* Xa = (float*)d_out;                 // A lives directly in d_out
    float* Xr = Xa + (size_t)nA * DD;          // R follows

    char* ws = (char*)d_ws;
    auto alloc = [&](size_t bytes) -> char* {
        char* p = ws; ws += (bytes + 255) & ~(size_t)255; return p;
    };
    __bf16* Wt  = (__bf16*)alloc((size_t)10 * 65536 * 2);
    __bf16* Abf = (__bf16*)alloc((size_t)(nA + SLACK) * DD * 2);
    __bf16* Rbf = (__bf16*)alloc((size_t)(nR + SLACK) * DD * 2);
    __bf16* qa  = (__bf16*)alloc((size_t)(nA + SLACK) * DD * 2);
    __bf16* ka  = (__bf16*)alloc((size_t)(nA + SLACK) * DD * 2);
    __bf16* qr  = (__bf16*)alloc((size_t)(nR + SLACK) * DD * 2);
    __bf16* kr  = (__bf16*)alloc((size_t)(nR + SLACK) * DD * 2);
    __bf16* vat = (__bf16*)alloc((size_t)DD * NB * AMAX * 2);   // [d][b*64 + tok]
    __bf16* vrt = (__bf16*)alloc((size_t)DD * NB * RMAX * 2);   // [d][b*512 + tok]

    WP wp;
    for (int i = 0; i < 10; ++i) wp.w[i] = (const float*)d_in[2 + i];
    // Wt slots: 0=W_lig_t 1=W_res_t 2=WQl 3=WKl 4=WVl 5=WOl 6=WQr 7=WKr 8=WVr 9=WOr

    wconv_kernel<<<10 * 256, 256, 0, stream>>>(wp, Wt);
    transform_kernel<<<(nA + 15) / 16, 128, 0, stream>>>(fatoms, Wt + 0 * 65536, Xa, Abf, nA);
    transform_kernel<<<(nR + 15) / 16, 128, 0, stream>>>(fres,   Wt + 1 * 65536, Xr, Rbf, nR);

    const long long vA = (long long)NB * AMAX;
    const long long vR = (long long)NB * RMAX;
    for (int it = 0; it < N_ITERS; ++it) {
        proj_kernel<<<dim3(NB, 3), 128, 0, stream>>>(
            Abf, Wt + 2 * 65536, Wt + 3 * 65536, Wt + 4 * 65536,
            qa, ka, vat, lig, AMAX, vA);
        proj_kernel<<<dim3(NB, 3), 128, 0, stream>>>(
            Rbf, Wt + 6 * 65536, Wt + 7 * 65536, Wt + 8 * 65536,
            qr, kr, vrt, res, RMAX, vR);
        // atoms attend residues -> update A (reads old-A projections only via qa)
        att_kernel<<<NB, 128, 0, stream>>>(qa, kr, vrt, Wt + 5 * 65536,
                                           Xa, Abf, lig, res, RMAX, vR);
        // residues attend atoms -> update R (ka/vat were built from pre-update A)
        att_kernel<<<NB, 128, 0, stream>>>(qr, ka, vat, Wt + 9 * 65536,
                                           Xr, Rbf, res, lig, AMAX, vA);
    }
    (void)n_in; (void)out_size; (void)ws_size;
}
